// Regression_53154515255830
// MI455X (gfx1250) — matrix operand order repaired and checked by reading, not compiled, not run on hardware
//
#include <hip/hip_runtime.h>
#include <hip/hip_bf16.h>

// ---------------------------------------------------------------------------
// Problem constants (from the reference)
// ---------------------------------------------------------------------------
#define BSZ   1024
#define TT    10
#define NN    (BSZ * TT)        // 10240 nodes
#define HH    256
#define EE    50
#define SEQL  50
#define NREGS 50
#define C3K   160               // conv3 K padded (150 -> 160)
#define C4K   512               // conv4 K = 128*4
#define L3    48                // conv3 output length
#define L4    45                // conv4 valid output length
#define NPB   16                // nodes per block in text CNN

// ---------------------------------------------------------------------------
// WMMA types / helpers (CDNA5 gfx1250, wave32)
// ---------------------------------------------------------------------------
typedef __bf16 v16bf __attribute__((ext_vector_type(16)));
typedef float  v8f   __attribute__((ext_vector_type(8)));
typedef unsigned int u32x4 __attribute__((ext_vector_type(4)));

union frag_cvt { u32x4 u[2]; v16bf v; };

__device__ inline v8f zero_v8f() {
    v8f z;
#pragma unroll
    for (int i = 0; i < 8; ++i) z[i] = 0.0f;
    return z;
}

// A-matrix fragment (16x32 bf16, M rows across lanes 0..15):
// lane element i holds K = (i&7) + 8*half + 16*(i>>3)
__device__ inline v16bf ldfragA(const __hip_bfloat16* p, int half) {
    frag_cvt c;
    c.u[0] = *reinterpret_cast<const u32x4*>(p + 8 * half);
    c.u[1] = *reinterpret_cast<const u32x4*>(p + 16 + 8 * half);
    return c.v;
}

// B-matrix fragment (32x16 bf16, N cols across lanes 0..15):
// lane element i holds K = (i&7) + 8*half + 16*(i>>3)
__device__ inline v16bf ldfragB(const __hip_bfloat16* p, int half) {
    frag_cvt c;
    const __hip_bfloat16* q = p;
    c.u[0] = *reinterpret_cast<const u32x4*>(q + 8 * half);
    c.u[1] = *reinterpret_cast<const u32x4*>(q + 16 + 8 * half);
    return c.v;
}

__device__ inline v8f wmma_bf16(v16bf a, v16bf b, v8f c) {
    return __builtin_amdgcn_wmma_f32_16x16x32_bf16(
        /*neg_a=*/false, a, /*neg_b=*/false, b,
        /*c_mod=*/(short)0, c, /*reuse_a=*/false, /*reuse_b=*/false);
}

__device__ inline __hip_bfloat16 f2bf(float x) { return __float2bfloat16(x); }
__device__ inline float bf2f(__hip_bfloat16 x) { return __bfloat162float(x); }

// CDNA5 async global->LDS copy (4 bytes per lane), tracked by ASYNCcnt.
// lds addr = low 32 bits of the generic shared pointer (LDS aperture).
__device__ inline void async_g2l_b32(void* lds_dst, const void* gsrc) {
    unsigned lds32 = (unsigned)(size_t)lds_dst;
    unsigned long long g64 = (unsigned long long)(size_t)gsrc;
    asm volatile("global_load_async_to_lds_b32 %0, %1, off"
                 :: "v"(lds32), "v"(g64) : "memory");
}
__device__ inline void async_wait0() {
    asm volatile("s_wait_asynccnt 0x0" ::: "memory");
}

// ---------------------------------------------------------------------------
// Weight conversion / padding kernels (f32 -> bf16 in workspace)
// ---------------------------------------------------------------------------
__global__ void f2bf_kernel(const float* __restrict__ src,
                            __hip_bfloat16* __restrict__ dst, int n) {
    int i = blockIdx.x * blockDim.x + threadIdx.x;
    if (i < n) dst[i] = f2bf(src[i]);
}

// conv3_w [128][50][3] -> [128][160] bf16, K padded with zeros
__global__ void pad_c3_kernel(const float* __restrict__ src,
                              __hip_bfloat16* __restrict__ dst) {
    int i = blockIdx.x * blockDim.x + threadIdx.x;
    if (i >= 128 * C3K) return;
    int co = i / C3K, kc = i - co * C3K;
    dst[i] = (kc < 150) ? f2bf(src[co * 150 + kc]) : f2bf(0.0f);
}

// reg_w [50][2560] -> [64][2560] bf16, rows padded with zeros
__global__ void pad_reg_kernel(const float* __restrict__ src,
                               __hip_bfloat16* __restrict__ dst) {
    int i = blockIdx.x * blockDim.x + threadIdx.x;
    if (i >= 64 * (TT * HH)) return;
    int j = i / (TT * HH);
    dst[i] = (j < NREGS) ? f2bf(src[i]) : f2bf(0.0f);
}

// ---------------------------------------------------------------------------
// Fused text CNN: async-gather-emb -> conv3 -> relu -> conv4 -> relu -> max
// One workgroup (8 waves) processes NPB nodes; all intermediates in LDS.
// LDS: Xs f32 [50][52] | B3t bf16 [48][160] | C3s bf16 [128][52] |
//      B4t bf16 [48][512]   = 88,224 bytes
// ---------------------------------------------------------------------------
__global__ __launch_bounds__(256) void textcnn_kernel(
    const int* __restrict__ text,           // [NN][50]
    const float* __restrict__ emb,          // [V][50]
    const __hip_bfloat16* __restrict__ w3,  // [128][160] bf16 (padded)
    const float* __restrict__ b3,           // [128]
    const __hip_bfloat16* __restrict__ w4,  // [128][512] bf16
    const float* __restrict__ b4,           // [128]
    __hip_bfloat16* __restrict__ txf)       // [NN][128] bf16
{
    extern __shared__ char smraw[];
    float* Xs           = (float*)smraw;                       // [50][52] f32
    __hip_bfloat16* bm  = (__hip_bfloat16*)(smraw + 50 * 52 * 4);
    __hip_bfloat16* B3t = bm;                                  // [48][160]
    __hip_bfloat16* C3s = bm + L3 * C3K;                       // [128][52]
    __hip_bfloat16* B4t = bm + L3 * C3K + 128 * 52;            // [48][512]

    const int tid  = threadIdx.x;
    const int lane = tid & 31;
    const int wave = tid >> 5;
    const int half = lane >> 4;
    const int l16  = lane & 15;

    for (int it = 0; it < NPB; ++it) {
        const int node = blockIdx.x * NPB + it;

        // phase 1: async gather (transpose): Xs[e][p] = emb[text[node,p]][e]
        for (int idx = tid; idx < SEQL * EE; idx += 256) {
            int p = idx / EE, e = idx - p * EE;   // consecutive threads: e contiguous
            int tok = text[node * SEQL + p];
            async_g2l_b32(&Xs[e * 52 + p], &emb[(size_t)tok * EE + e]);
        }
        async_wait0();
        __syncthreads();

        // phase 2: im2col for conv3:  B3t[p][ci*3+k] = bf16(Xs[ci][p+k])
        for (int idx = tid; idx < L3 * C3K; idx += 256) {
            int p = idx / C3K, kc = idx - p * C3K;
            __hip_bfloat16 v;
            if (kc < 150) { int ci = kc / 3; int k = kc - ci * 3; v = f2bf(Xs[ci * 52 + p + k]); }
            else v = f2bf(0.0f);
            B3t[p * C3K + kc] = v;
        }
        __syncthreads();

        // phase 3: conv3 GEMM via WMMA, wave w owns co rows [16w,16w+16);
        // A fragment reused across the 3 position tiles.
        {
            const __hip_bfloat16* arow = w3 + (size_t)(16 * wave + l16) * C3K;
            const __hip_bfloat16* brow = B3t + l16 * C3K;
            v8f acc0 = zero_v8f(), acc1 = zero_v8f(), acc2 = zero_v8f();
#pragma unroll
            for (int k0 = 0; k0 < C3K; k0 += 32) {
                v16bf af = ldfragA(arow + k0, half);
                acc0 = wmma_bf16(af, ldfragB(brow + 0 * 16 * C3K + k0, half), acc0);
                acc1 = wmma_bf16(af, ldfragB(brow + 1 * 16 * C3K + k0, half), acc1);
                acc2 = wmma_bf16(af, ldfragB(brow + 2 * 16 * C3K + k0, half), acc2);
            }
            v8f accs[3] = {acc0, acc1, acc2};
#pragma unroll
            for (int nt = 0; nt < 3; ++nt) {
#pragma unroll
                for (int r = 0; r < 8; ++r) {
                    int co = 16 * wave + r + 8 * half;
                    int p  = nt * 16 + l16;
                    float v = fmaxf(accs[nt][r] + b3[co], 0.0f);  // bias + relu
                    C3s[co * 52 + p] = f2bf(v);
                }
            }
        }
        __syncthreads();

        // phase 4: im2col for conv4:  B4t[p][ci*4+k] = C3s[ci][p+k] (0 past col 47)
        for (int idx = tid; idx < L3 * C4K; idx += 256) {
            int p = idx >> 9, kc = idx & 511;
            int ci = kc >> 2, k = kc & 3;
            int c = p + k;
            B4t[p * C4K + kc] = (c < L3) ? C3s[ci * 52 + c] : f2bf(0.0f);
        }
        __syncthreads();

        // phase 5: conv4 GEMM + bias + relu + masked max over positions
        {
            const __hip_bfloat16* arow = w4 + (size_t)(16 * wave + l16) * C4K;
            const __hip_bfloat16* brow = B4t + l16 * C4K;
            v8f acc0 = zero_v8f(), acc1 = zero_v8f(), acc2 = zero_v8f();
#pragma unroll
            for (int k0 = 0; k0 < C4K; k0 += 32) {
                v16bf af = ldfragA(arow + k0, half);
                acc0 = wmma_bf16(af, ldfragB(brow + 0 * 16 * C4K + k0, half), acc0);
                acc1 = wmma_bf16(af, ldfragB(brow + 1 * 16 * C4K + k0, half), acc1);
                acc2 = wmma_bf16(af, ldfragB(brow + 2 * 16 * C4K + k0, half), acc2);
            }
            v8f accs[3] = {acc0, acc1, acc2};
            float vmax[8];
#pragma unroll
            for (int r = 0; r < 8; ++r) vmax[r] = 0.0f;  // relu>=0: 0 is identity
#pragma unroll
            for (int nt = 0; nt < 3; ++nt) {
                int p = nt * 16 + l16;
                bool valid = (p < L4);
#pragma unroll
                for (int r = 0; r < 8; ++r) {
                    int co = 16 * wave + r + 8 * half;
                    float v = fmaxf(accs[nt][r] + b4[co], 0.0f);
                    if (valid) vmax[r] = fmaxf(vmax[r], v);
                }
            }
            // reduce max across the 16 lanes of each half
#pragma unroll
            for (int off = 1; off < 16; off <<= 1)
#pragma unroll
                for (int r = 0; r < 8; ++r)
                    vmax[r] = fmaxf(vmax[r], __shfl_xor(vmax[r], off, 32));
            if (l16 == 0) {
#pragma unroll
                for (int r = 0; r < 8; ++r)
                    txf[(size_t)node * 128 + 16 * wave + r + 8 * half] = f2bf(vmax[r]);
            }
        }
        __syncthreads();
    }
}

// ---------------------------------------------------------------------------
// Generic WMMA GEMM:  D[i][j] = act( sum_k A[i][k] * W[j][k] + bias[j] )
// A: bf16 [M][lda], W: bf16 [Ncols][ldw].  Each wave owns a 16x64 output
// tile (one A fragment feeds 4 WMMAs per k-step).  Output f32 or bf16 with
// column offset + valid-column mask.
// grid = (Ncols/64, M/128), block = 256 (8 waves)
// ---------------------------------------------------------------------------
__global__ __launch_bounds__(256) void gemm_wmma_kernel(
    const __hip_bfloat16* __restrict__ A, int lda,
    const __hip_bfloat16* __restrict__ W, int ldw,
    const float* __restrict__ bias,
    float* __restrict__ f32out, __hip_bfloat16* __restrict__ bf16out,
    int ldo, int col_off, int nvalid, int K, int act)
{
    const int lane  = threadIdx.x & 31;
    const int wave  = threadIdx.x >> 5;
    const int half  = lane >> 4;
    const int l16   = lane & 15;
    const int nbase = blockIdx.x * 4;            // first of 4 N-tiles
    const int mtile = blockIdx.y * 8 + wave;

    const __hip_bfloat16* arow = A + (size_t)(mtile * 16 + l16) * lda;
    const __hip_bfloat16* wrow = W + (size_t)(nbase * 16 + l16) * ldw;

    v8f acc0 = zero_v8f(), acc1 = zero_v8f(), acc2 = zero_v8f(), acc3 = zero_v8f();
    for (int k0 = 0; k0 < K; k0 += 32) {
        v16bf af = ldfragA(arow + k0, half);
        acc0 = wmma_bf16(af, ldfragB(wrow + (size_t)0 * 16 * ldw + k0, half), acc0);
        acc1 = wmma_bf16(af, ldfragB(wrow + (size_t)1 * 16 * ldw + k0, half), acc1);
        acc2 = wmma_bf16(af, ldfragB(wrow + (size_t)2 * 16 * ldw + k0, half), acc2);
        acc3 = wmma_bf16(af, ldfragB(wrow + (size_t)3 * 16 * ldw + k0, half), acc3);
    }
    v8f accs[4] = {acc0, acc1, acc2, acc3};

#pragma unroll
    for (int j = 0; j < 4; ++j) {
        const int col = (nbase + j) * 16 + l16;
        if (col >= nvalid) continue;
        const float bv = bias[col];
#pragma unroll
        for (int r = 0; r < 8; ++r) {
            int row = mtile * 16 + r + 8 * half;
            float v = accs[j][r] + bv;
            if (act) v = fmaxf(v, 0.0f);
            size_t o = (size_t)row * ldo + col_off + col;
            if (f32out) f32out[o] = v;
            else        bf16out[o] = f2bf(v);
        }
    }
}

// ---------------------------------------------------------------------------
// Per-node scalar feature projections -> F[:, 0:768]  (bf16)
// ---------------------------------------------------------------------------
__global__ void feat_kernel(const float* __restrict__ lags,
                            const float* __restrict__ weather,
                            const float* __restrict__ event,
                            const float* __restrict__ lw, const float* __restrict__ lb,
                            const float* __restrict__ ww, const float* __restrict__ wb,
                            const float* __restrict__ ew, const float* __restrict__ eb,
                            __hip_bfloat16* __restrict__ F)
{
    int idx = blockIdx.x * blockDim.x + threadIdx.x;
    if (idx >= NN * 768) return;
    int n = idx / 768, c = idx - n * 768;
    float v;
    if (c < 256) {
        v = lags[n] * lw[c] + lb[c];
    } else if (c < 512) {
        int h = c - 256;
        v = weather[n * 2 + 0] * ww[h * 2 + 0] + weather[n * 2 + 1] * ww[h * 2 + 1] + wb[h];
    } else {
        int h = c - 512;
        v = eb[h];
#pragma unroll
        for (int i = 0; i < 4; ++i) v += event[n * 4 + i] * ew[h * 4 + i];
    }
    F[(size_t)n * 1024 + c] = f2bf(v);
}

// ---------------------------------------------------------------------------
// Graph aggregation (fixed 10-node window graph, in-degree <= 2)
// ---------------------------------------------------------------------------
__global__ void agg_kernel(const __hip_bfloat16* __restrict__ F,
                           __hip_bfloat16* __restrict__ hcat)
{
    int idx = blockIdx.x * blockDim.x + threadIdx.x;
    if (idx >= NN * 1024) return;
    int n = idx >> 10, c = idx & 1023;
    int g = n / 10, d = n - g * 10;
    int s0 = (d == 0) ? 9 : ((d == 1) ? 0 : d - 2);
    float v = bf2f(F[(size_t)(g * 10 + s0) * 1024 + c]);
    if (d >= 2) v += bf2f(F[(size_t)(g * 10 + d - 1) * 1024 + c]);
    hcat[idx] = f2bf(v);
}

// ---------------------------------------------------------------------------
// BatchNorm batch statistics over the merge output (one block per channel)
// a[c] = gamma/sqrt(var+eps), b[c] = beta - mu*a
// ---------------------------------------------------------------------------
__global__ void bn_stats_kernel(const __hip_bfloat16* __restrict__ m,
                                const float* __restrict__ gamma,
                                const float* __restrict__ beta,
                                float* __restrict__ avec, float* __restrict__ bvec)
{
    __shared__ float ssum[256], ssq[256];
    int c = blockIdx.x, t = threadIdx.x;
    float s = 0.0f, q = 0.0f;
    for (int r = t; r < NN; r += 256) {
        float v = bf2f(m[(size_t)r * HH + c]);
        s += v; q += v * v;
    }
    ssum[t] = s; ssq[t] = q;
    __syncthreads();
    for (int off = 128; off > 0; off >>= 1) {
        if (t < off) { ssum[t] += ssum[t + off]; ssq[t] += ssq[t + off]; }
        __syncthreads();
    }
    if (t == 0) {
        float mu = ssum[0] / (float)NN;
        float var = ssq[0] / (float)NN - mu * mu;
        float a = gamma[c] * rsqrtf(var + 1e-5f);
        avec[c] = a;
        bvec[c] = beta[c] - mu * a;
    }
}

// Fold BN affine into mh weights: w'[j][k] = mh_w[j][k]*a[k] (bf16),
// bias2[j] = mh_b[j] + sum_k mh_w[j][k]*b[k]
__global__ void fold_mh_kernel(const float* __restrict__ mh_w,
                               const float* __restrict__ mh_b,
                               const float* __restrict__ avec,
                               const float* __restrict__ bvec,
                               __hip_bfloat16* __restrict__ w_mh,
                               float* __restrict__ bias2)
{
    __shared__ float red[256];
    int j = blockIdx.x, k = threadIdx.x;
    float w = mh_w[j * HH + k];
    w_mh[j * HH + k] = f2bf(w * avec[k]);
    red[k] = w * bvec[k];
    __syncthreads();
    for (int off = 128; off > 0; off >>= 1) {
        if (k < off) red[k] += red[k + off];
        __syncthreads();
    }
    if (k == 0) bias2[j] = mh_b[j] + red[0];
}

// ---------------------------------------------------------------------------
// GRU
// ---------------------------------------------------------------------------
__global__ void zero_h_kernel(float* __restrict__ Hf, __hip_bfloat16* __restrict__ Hb)
{
    int i = blockIdx.x * blockDim.x + threadIdx.x;
    if (i < BSZ * HH) { Hf[i] = 0.0f; Hb[i] = f2bf(0.0f); }
}

__global__ void gru_gate_kernel(const float* __restrict__ GI,   // [NN][768]
                                const float* __restrict__ GH,   // [BSZ][768]
                                float* __restrict__ Hf,         // [BSZ][256]
                                __hip_bfloat16* __restrict__ Hb,
                                __hip_bfloat16* __restrict__ Hflat, // [BSZ][2560]
                                int t)
{
    int idx = blockIdx.x * blockDim.x + threadIdx.x;
    if (idx >= BSZ * HH) return;
    int g = idx >> 8, j = idx & 255;
    const float* gi = GI + (size_t)(g * TT + t) * 768;
    const float* gh = GH + (size_t)g * 768;
    float r  = 1.0f / (1.0f + __expf(-(gi[j]       + gh[j])));
    float z  = 1.0f / (1.0f + __expf(-(gi[256 + j] + gh[256 + j])));
    float nn = tanhf(gi[512 + j] + r * gh[512 + j]);
    float hp = Hf[idx];
    float h  = (1.0f - z) * nn + z * hp;
    Hf[idx] = h;
    Hb[idx] = f2bf(h);
    Hflat[(size_t)g * (TT * HH) + t * HH + j] = f2bf(h);
}

// ---------------------------------------------------------------------------
// Host-side launch
// ---------------------------------------------------------------------------
extern "C" void kernel_launch(void* const* d_in, const int* in_sizes, int n_in,
                              void* d_out, int out_size, void* d_ws, size_t ws_size,
                              hipStream_t stream) {
    (void)in_sizes; (void)n_in; (void)out_size; (void)ws_size;

    const float* lags      = (const float*)d_in[0];
    const float* weather   = (const float*)d_in[1];
    const float* event     = (const float*)d_in[2];
    const int*   text      = (const int*)  d_in[3];
    const float* emb       = (const float*)d_in[4];
    const float* conv3_w   = (const float*)d_in[5];
    const float* conv3_b   = (const float*)d_in[6];
    const float* conv4_w   = (const float*)d_in[7];
    const float* conv4_b   = (const float*)d_in[8];
    const float* lags_w    = (const float*)d_in[9];
    const float* lags_b    = (const float*)d_in[10];
    const float* weather_w = (const float*)d_in[11];
    const float* weather_b = (const float*)d_in[12];
    const float* event_w   = (const float*)d_in[13];
    const float* event_b   = (const float*)d_in[14];
    const float* text_w    = (const float*)d_in[15];
    const float* text_b    = (const float*)d_in[16];
    const float* merge_w   = (const float*)d_in[17];
    const float* merge_b   = (const float*)d_in[18];
    const float* bn_gamma  = (const float*)d_in[19];
    const float* bn_beta   = (const float*)d_in[20];
    const float* mh_w      = (const float*)d_in[21];
    const float* mh_b      = (const float*)d_in[22];
    const float* gru_wih   = (const float*)d_in[23];
    const float* gru_whh   = (const float*)d_in[24];
    const float* gru_bih   = (const float*)d_in[25];
    const float* gru_bhh   = (const float*)d_in[26];
    const float* reg_w     = (const float*)d_in[27];
    const float* reg_b     = (const float*)d_in[28];
    float* out = (float*)d_out;

    // ----- workspace carve-up (256B aligned regions) -----
    char* ws = (char*)d_ws;
    size_t off = 0;
    auto carve = [&](size_t bytes) -> char* {
        char* p = ws + off;
        off = (off + bytes + 255) & ~(size_t)255;
        return p;
    };
    __hip_bfloat16* txf     = (__hip_bfloat16*)carve((size_t)NN * 128 * 2);
    __hip_bfloat16* F       = (__hip_bfloat16*)carve((size_t)NN * 1024 * 2);
    __hip_bfloat16* hcat    = (__hip_bfloat16*)carve((size_t)NN * 1024 * 2);
    __hip_bfloat16* m_bf    = (__hip_bfloat16*)carve((size_t)NN * HH * 2);
    __hip_bfloat16* xmh_bf  = (__hip_bfloat16*)carve((size_t)NN * HH * 2);
    float*          GI      = (float*)carve((size_t)NN * 768 * 4);
    float*          GH      = (float*)carve((size_t)BSZ * 768 * 4);
    float*          Hf      = (float*)carve((size_t)BSZ * HH * 4);
    __hip_bfloat16* Hb      = (__hip_bfloat16*)carve((size_t)BSZ * HH * 2);
    __hip_bfloat16* Hflat   = (__hip_bfloat16*)carve((size_t)BSZ * TT * HH * 2);
    __hip_bfloat16* w_c3    = (__hip_bfloat16*)carve((size_t)128 * C3K * 2);
    __hip_bfloat16* w_c4    = (__hip_bfloat16*)carve((size_t)128 * C4K * 2);
    __hip_bfloat16* w_merge = (__hip_bfloat16*)carve((size_t)HH * 1024 * 2);
    __hip_bfloat16* w_mh    = (__hip_bfloat16*)carve((size_t)HH * HH * 2);
    __hip_bfloat16* w_gih   = (__hip_bfloat16*)carve((size_t)768 * HH * 2);
    __hip_bfloat16* w_ghh   = (__hip_bfloat16*)carve((size_t)768 * HH * 2);
    __hip_bfloat16* w_reg   = (__hip_bfloat16*)carve((size_t)64 * TT * HH * 2);
    __hip_bfloat16* w_text  = (__hip_bfloat16*)carve((size_t)HH * 128 * 2);
    float*          avec    = (float*)carve(HH * 4);
    float*          bvec    = (float*)carve(HH * 4);
    float*          bias2   = (float*)carve(HH * 4);

    const int BLK = 256;
    auto nblk = [](size_t n, int b) { return (unsigned)((n + b - 1) / b); };

    // ----- weight conversion -----
    pad_c3_kernel<<<nblk(128 * C3K, BLK), BLK, 0, stream>>>(conv3_w, w_c3);
    f2bf_kernel<<<nblk(128 * C4K, BLK), BLK, 0, stream>>>(conv4_w, w_c4, 128 * C4K);
    f2bf_kernel<<<nblk(HH * 1024, BLK), BLK, 0, stream>>>(merge_w, w_merge, HH * 1024);
    f2bf_kernel<<<nblk(768 * HH, BLK), BLK, 0, stream>>>(gru_wih, w_gih, 768 * HH);
    f2bf_kernel<<<nblk(768 * HH, BLK), BLK, 0, stream>>>(gru_whh, w_ghh, 768 * HH);
    f2bf_kernel<<<nblk(HH * 128, BLK), BLK, 0, stream>>>(text_w, w_text, HH * 128);
    pad_reg_kernel<<<nblk(64 * TT * HH, BLK), BLK, 0, stream>>>(reg_w, w_reg);

    // ----- fused text CNN (88 KB dynamic LDS) -----
    const int smem = 50 * 52 * 4 + (L3 * C3K + 128 * 52 + L3 * C4K) * 2;  // 88,224
    (void)hipFuncSetAttribute((const void*)textcnn_kernel,
                              hipFuncAttributeMaxDynamicSharedMemorySize, smem);
    textcnn_kernel<<<NN / NPB, BLK, smem, stream>>>(text, emb, w_c3, conv3_b,
                                                    w_c4, conv4_b, txf);

    // ----- per-node features: F[:,0:768] scalar branches -----
    feat_kernel<<<nblk((size_t)NN * 768, BLK), BLK, 0, stream>>>(
        lags, weather, event, lags_w, lags_b, weather_w, weather_b,
        event_w, event_b, F);

    // ----- text linear -> F[:,768:1024]   (M=NN, N=256, K=128) -----
    gemm_wmma_kernel<<<dim3(HH / 64, NN / 128), BLK, 0, stream>>>(
        txf, 128, w_text, 128, text_b, nullptr, F, 1024, 768, 1 << 28, 128, 0);

    // ----- graph aggregation -----
    agg_kernel<<<nblk((size_t)NN * 1024, BLK), BLK, 0, stream>>>(F, hcat);

    // ----- merge GEMM + relu   (M=NN, N=256, K=1024) -----
    gemm_wmma_kernel<<<dim3(HH / 64, NN / 128), BLK, 0, stream>>>(
        hcat, 1024, w_merge, 1024, merge_b, nullptr, m_bf, HH, 0, 1 << 28, 1024, 1);

    // ----- batch-norm stats + fold into mh -----
    bn_stats_kernel<<<HH, BLK, 0, stream>>>(m_bf, bn_gamma, bn_beta, avec, bvec);
    fold_mh_kernel<<<HH, BLK, 0, stream>>>(mh_w, mh_b, avec, bvec, w_mh, bias2);

    // ----- mh GEMM (BN folded)   (M=NN, N=256, K=256) -----
    gemm_wmma_kernel<<<dim3(HH / 64, NN / 128), BLK, 0, stream>>>(
        m_bf, HH, w_mh, HH, bias2, nullptr, xmh_bf, HH, 0, 1 << 28, HH, 0);

    // ----- GRU input gates for all steps   (M=NN, N=768, K=256) -----
    gemm_wmma_kernel<<<dim3(768 / 64, NN / 128), BLK, 0, stream>>>(
        xmh_bf, HH, w_gih, HH, gru_bih, GI, nullptr, 768, 0, 1 << 28, HH, 0);

    // ----- GRU recurrence -----
    zero_h_kernel<<<nblk(BSZ * HH, BLK), BLK, 0, stream>>>(Hf, Hb);
    for (int t = 0; t < TT; ++t) {
        // gh = h_{t-1} @ whh^T + bhh   (M=1024, N=768, K=256)
        gemm_wmma_kernel<<<dim3(768 / 64, BSZ / 128), BLK, 0, stream>>>(
            Hb, HH, w_ghh, HH, gru_bhh, GH, nullptr, 768, 0, 1 << 28, HH, 0);
        gru_gate_kernel<<<nblk(BSZ * HH, BLK), BLK, 0, stream>>>(GI, GH, Hf, Hb, Hflat, t);
    }

    // ----- regression head   (M=1024, N=64 padded -> 50 valid, K=2560) -----
    gemm_wmma_kernel<<<dim3(64 / 64, BSZ / 128), BLK, 0, stream>>>(
        Hflat, TT * HH, w_reg, TT * HH, reg_b, out, nullptr, NREGS, 0, NREGS,
        TT * HH, 0);
}
